// SDM_35210141892755
// MI455X (gfx1250) — matrix operand order repaired and checked by reading, not compiled, not run on hardware
//
#include <hip/hip_runtime.h>
#include <math.h>

typedef _Float16 h16_t;
typedef _Float16 v16h __attribute__((ext_vector_type(16)));
typedef _Float16 v8h  __attribute__((ext_vector_type(8)));
typedef float    v8f  __attribute__((ext_vector_type(8)));

#define NTOK 24576      // 2 batches * 4 dirs * 3072 tokens
#define LSEQ 3072
#define CS   32         // scan chunk size
#define NC   96         // chunks per sequence (CS*NC == LSEQ)

// ---------------- weight convert (f32 -> f16, optional zero row pad) --------
__global__ void k_cvt(const float* __restrict__ src, h16_t* __restrict__ dst,
                      int rows, int cols, int src_rows) {
    int i = blockIdx.x * 256 + threadIdx.x;
    if (i >= rows * cols) return;
    int r = i / cols;
    dst[i] = (r < src_rows) ? (h16_t)src[i] : (h16_t)0.0f;
}

// ---------------- gather / merge index mapping ------------------------------
__device__ __forceinline__ int map_src(int d, int r, int c, int b) {
    int hh, ww;
    if (d == 0)      { hh = 2 * (r >> 6);     ww = 2 * (r & 63); }
    else if (d == 1) { ww = 2 * (r / 48);     hh = 2 * (r % 48) + 1; }
    else if (d == 2) { hh = 2 * (r >> 6);     ww = 2 * (r & 63) + 1; }
    else             { ww = 2 * (r / 48) + 1; hh = 2 * (r % 48) + 1; }
    return ((b * 128 + c) * 96 + hh) * 128 + ww;
}

__global__ void k_gather(const float* __restrict__ f, float* __restrict__ xs) {
    int i = blockIdx.x * 256 + threadIdx.x;
    const int CL = 128 * LSEQ;
    if (i >= 8 * CL) return;
    int sq = i / CL, idx = i - sq * CL;
    int b = sq >> 2, d = sq & 3;
    int c = idx / LSEQ, r = idx - c * LSEQ;
    xs[i] = f[map_src(d, r, c, b)];
}

__global__ void k_merge(const float* __restrict__ y, float* __restrict__ out) {
    int i = blockIdx.x * 256 + threadIdx.x;
    const int CL = 128 * LSEQ;
    if (i >= 8 * CL) return;
    int sq = i / CL, idx = i - sq * CL;
    int b = sq >> 2, d = sq & 3;
    int c = idx / LSEQ, r = idx - c * LSEQ;
    out[map_src(d, r, c, b)] = y[i];
}

// ---------------- LayerNorm (one wave per token, 4 vals/lane) --> f16 -------
__global__ __launch_bounds__(128)
void k_ln(const float* __restrict__ x, const float* __restrict__ w,
          const float* __restrict__ bb, h16_t* __restrict__ out, int ntok) {
    int wv = (blockIdx.x * blockDim.x + threadIdx.x) >> 5;
    int lane = threadIdx.x & 31;
    if (wv >= ntok) return;
    const float* row = x + (size_t)wv * 128;
    float v[4], s = 0.f;
#pragma unroll
    for (int i = 0; i < 4; i++) { v[i] = row[lane * 4 + i]; s += v[i]; }
#pragma unroll
    for (int m = 16; m >= 1; m >>= 1) s += __shfl_xor(s, m, 32);
    float mean = s * (1.f / 128.f), q = 0.f;
#pragma unroll
    for (int i = 0; i < 4; i++) { float dd = v[i] - mean; q += dd * dd; }
#pragma unroll
    for (int m = 16; m >= 1; m >>= 1) q += __shfl_xor(q, m, 32);
    float rstd = rsqrtf(q * (1.f / 128.f) + 1e-5f);
#pragma unroll
    for (int i = 0; i < 4; i++) {
        int cc = lane * 4 + i;
        out[(size_t)wv * 128 + cc] = (h16_t)((v[i] - mean) * rstd * w[cc] + bb[cc]);
    }
}

// ---------------- final LN + residual --> f32 -------------------------------
__global__ __launch_bounds__(128)
void k_lnres(const float* __restrict__ x, const float* __restrict__ w,
             const float* __restrict__ bb, const float* __restrict__ resid,
             float* __restrict__ out, int ntok) {
    int wv = (blockIdx.x * blockDim.x + threadIdx.x) >> 5;
    int lane = threadIdx.x & 31;
    if (wv >= ntok) return;
    const float* row = x + (size_t)wv * 128;
    float v[4], s = 0.f;
#pragma unroll
    for (int i = 0; i < 4; i++) { v[i] = row[lane * 4 + i]; s += v[i]; }
#pragma unroll
    for (int m = 16; m >= 1; m >>= 1) s += __shfl_xor(s, m, 32);
    float mean = s * (1.f / 128.f), q = 0.f;
#pragma unroll
    for (int i = 0; i < 4; i++) { float dd = v[i] - mean; q += dd * dd; }
#pragma unroll
    for (int m = 16; m >= 1; m >>= 1) q += __shfl_xor(q, m, 32);
    float rstd = rsqrtf(q * (1.f / 128.f) + 1e-5f);
#pragma unroll
    for (int i = 0; i < 4; i++) {
        int cc = lane * 4 + i;
        out[(size_t)wv * 128 + cc] =
            (v[i] - mean) * rstd * w[cc] + bb[cc] + resid[(size_t)wv * 128 + cc];
    }
}

// ---------------- WMMA GEMM:  out[M,N] = act(A[M,K] @ W[N,K]^T) (+resid) ----
// One wave per 16-row tile; NT (compile-time) 16-col tiles per wave.
// No dynamic guards in the K loop -> straight-line load/load/wmma code.
template <int NT, bool HASF, bool HAS16, bool HASR, int ACT>
__global__ __launch_bounds__(32)
void k_gemm(const h16_t* __restrict__ A, const h16_t* __restrict__ W,
            int M, int K,
            float* __restrict__ outF, int ldF,
            h16_t* __restrict__ out16, int ld16,
            const float* __restrict__ resid, int ldR) {
    int lane = threadIdx.x;
    int mt = blockIdx.x;
    int nb = blockIdx.y * (NT * 16);
    int half = lane >> 4;
    int l15 = lane & 15;
    const h16_t* arow = A + (size_t)(mt * 16 + l15) * K + half * 16;

    v8f acc[NT];
#pragma unroll
    for (int t = 0; t < NT; t++) acc[t] = v8f{0};

    for (int kk = 0; kk < K; kk += 32) {
        v8h alo = *(const v8h*)(arow + kk);
        v8h ahi = *(const v8h*)(arow + kk + 8);
        v16h a;
#pragma unroll
        for (int i = 0; i < 8; i++) { a[i] = alo[i]; a[i + 8] = ahi[i]; }
#pragma unroll
        for (int t = 0; t < NT; t++) {
            int ncol = nb + t * 16 + l15;
            v16h b = *(const v16h*)(W + (size_t)ncol * K + kk + half * 16);
            acc[t] = __builtin_amdgcn_wmma_f32_16x16x32_f16(
                false, a, false, b, (short)0, acc[t], false, false);
        }
    }
#pragma unroll
    for (int t = 0; t < NT; t++) {
        int ncol = nb + t * 16 + l15;
#pragma unroll
        for (int v = 0; v < 8; v++) {
            int r = mt * 16 + half * 8 + v;
            float c = acc[t][v];
            if (ACT == 1) c = 0.5f * c * (1.0f + erff(c * 0.70710678f)); // exact GELU
            if (HASR) c += resid[(size_t)r * ldR + ncol];
            if (HASF) outF[(size_t)r * ldF + ncol] = c;
            if (HAS16) out16[(size_t)r * ld16 + ncol] = (h16_t)c;
        }
    }
}

// ---------------- causal depthwise conv (k=4) + SiLU ------------------------
__global__ void k_conv(const h16_t* __restrict__ xz, const float* __restrict__ cw,
                       const float* __restrict__ cb, h16_t* __restrict__ xo, int sq) {
    int i = blockIdx.x * 256 + threadIdx.x;
    if (i >= LSEQ * 256) return;
    int l = i >> 8, d = i & 255;
    float s = cb[d];
#pragma unroll
    for (int j = 0; j < 4; j++) {
        int l2 = l - 3 + j;
        if (l2 >= 0)
            s += cw[d * 4 + j] * (float)xz[(size_t)(sq * LSEQ + l2) * 512 + d];
    }
    float sl = s / (1.f + __expf(-s));
    xo[(size_t)(sq * LSEQ + l) * 256 + d] = (h16_t)sl;
}

__device__ __forceinline__ float softplusf(float t) {
    return (t > 20.f) ? t : log1pf(__expf(t));
}

// ---------------- scan pass A: per-chunk (prod a, local state) --------------
__global__ __launch_bounds__(256)
void k_scanA(const float* __restrict__ xdbl, const h16_t* __restrict__ x16,
             const float* __restrict__ A_log, const float* __restrict__ dtw,
             const float* __restrict__ dtb, float* __restrict__ chA,
             float* __restrict__ chB, int sq) {
    int d = threadIdx.x, c = blockIdx.x;
    float Av[16], wv[8], h[16], P[16];
#pragma unroll
    for (int n = 0; n < 16; n++) { Av[n] = -__expf(A_log[d * 16 + n]); h[n] = 0.f; P[n] = 1.f; }
#pragma unroll
    for (int r = 0; r < 8; r++) wv[r] = dtw[d * 8 + r];
    float bv = dtb[d];
    int t0 = sq * LSEQ + c * CS;
    for (int l = 0; l < CS; l++) {
        const float* row = xdbl + (size_t)(t0 + l) * 48;
        float dt = bv;
#pragma unroll
        for (int r = 0; r < 8; r++) dt += wv[r] * row[r];
        dt = softplusf(dt);
        float dx = dt * (float)x16[(size_t)(t0 + l) * 256 + d];
#pragma unroll
        for (int n = 0; n < 16; n++) {
            float a = __expf(dt * Av[n]);
            h[n] = a * h[n] + dx * row[8 + n];
            P[n] *= a;
        }
    }
#pragma unroll
    for (int n = 0; n < 16; n++) {
        chA[c * 4096 + d * 16 + n] = P[n];
        chB[c * 4096 + d * 16 + n] = h[n];
    }
}

// ---------------- scan pass B: chain chunk states per (d,n) -----------------
__global__ __launch_bounds__(256)
void k_scanB(const float* __restrict__ chA, const float* __restrict__ chB,
             float* __restrict__ hinit) {
    int i = blockIdx.x * 256 + threadIdx.x;
    if (i >= 4096) return;
    float h = 0.f;
    for (int c = 0; c < NC; c++) {
        hinit[c * 4096 + i] = h;
        h = chA[c * 4096 + i] * h + chB[c * 4096 + i];
    }
}

// ---------------- scan pass C: replay with correct h_in, emit y -------------
__global__ __launch_bounds__(256)
void k_scanC(const float* __restrict__ xdbl, const h16_t* __restrict__ x16,
             const float* __restrict__ A_log, const float* __restrict__ dtw,
             const float* __restrict__ dtb, const float* __restrict__ hinit,
             h16_t* __restrict__ y16, int sq) {
    int d = threadIdx.x, c = blockIdx.x;
    float Av[16], wv[8], h[16];
#pragma unroll
    for (int n = 0; n < 16; n++) { Av[n] = -__expf(A_log[d * 16 + n]); h[n] = hinit[c * 4096 + d * 16 + n]; }
#pragma unroll
    for (int r = 0; r < 8; r++) wv[r] = dtw[d * 8 + r];
    float bv = dtb[d];
    int t0 = sq * LSEQ + c * CS;
    for (int l = 0; l < CS; l++) {
        const float* row = xdbl + (size_t)(t0 + l) * 48;
        float dt = bv;
#pragma unroll
        for (int r = 0; r < 8; r++) dt += wv[r] * row[r];
        dt = softplusf(dt);
        float dx = dt * (float)x16[(size_t)(t0 + l) * 256 + d];
        float y = 0.f;
#pragma unroll
        for (int n = 0; n < 16; n++) {
            float a = __expf(dt * Av[n]);
            h[n] = a * h[n] + dx * row[8 + n];
            y += h[n] * row[24 + n];
        }
        y16[(size_t)(t0 + l) * 256 + d] = (h16_t)y;
    }
}

// ---------------- y = (y + x*D) * silu(z) -----------------------------------
__global__ void k_post(h16_t* __restrict__ y16, const h16_t* __restrict__ x16,
                       const h16_t* __restrict__ xz16, const float* __restrict__ Dp,
                       int sq) {
    int i = blockIdx.x * 256 + threadIdx.x;
    if (i >= LSEQ * 256) return;
    int l = i >> 8, d = i & 255;
    size_t t = (size_t)(sq * LSEQ + l);
    float y = (float)y16[t * 256 + d] + (float)x16[t * 256 + d] * Dp[d];
    float z = (float)xz16[t * 512 + 256 + d];
    float sz = z / (1.f + __expf(-z));
    y16[t * 256 + d] = (h16_t)(y * sz);
}

// ============================================================================
extern "C" void kernel_launch(void* const* d_in, const int* in_sizes, int n_in,
                              void* d_out, int out_size, void* d_ws, size_t ws_size,
                              hipStream_t stream) {
    // ---- input mapping (auto-detect insertion vs sorted pytree order) ----
    const float *feature, *mlp_w1, *mlp_w2, *norm_w, *norm_b;
    const float *in_proj[4], *conv_w[4], *conv_b[4], *x_proj[4], *dt_w[4], *dt_b[4],
                *A_log[4], *Dp[4], *out_proj[4], *ln_w[4], *ln_b[4];
    bool insertion = (n_in > 1 && in_sizes[1] == 1);
    feature = (const float*)d_in[0];
    if (insertion) {
        for (int l = 0; l < 4; l++) {
            int b = 2 + l * 11;
            in_proj[l] = (const float*)d_in[b + 0]; conv_w[l] = (const float*)d_in[b + 1];
            conv_b[l] = (const float*)d_in[b + 2];  x_proj[l] = (const float*)d_in[b + 3];
            dt_w[l] = (const float*)d_in[b + 4];    dt_b[l] = (const float*)d_in[b + 5];
            A_log[l] = (const float*)d_in[b + 6];   Dp[l] = (const float*)d_in[b + 7];
            out_proj[l] = (const float*)d_in[b + 8];
            ln_w[l] = (const float*)d_in[b + 9];    ln_b[l] = (const float*)d_in[b + 10];
        }
        mlp_w1 = (const float*)d_in[46]; mlp_w2 = (const float*)d_in[47];
        norm_w = (const float*)d_in[48]; norm_b = (const float*)d_in[49];
    } else { // sorted keys: A_log,D,conv_b,conv_w,dt_b,dt_w,in_proj,ln_b,ln_w,out_proj,x_proj
        for (int l = 0; l < 4; l++) {
            int b = 1 + l * 11;
            A_log[l] = (const float*)d_in[b + 0];   Dp[l] = (const float*)d_in[b + 1];
            conv_b[l] = (const float*)d_in[b + 2];  conv_w[l] = (const float*)d_in[b + 3];
            dt_b[l] = (const float*)d_in[b + 4];    dt_w[l] = (const float*)d_in[b + 5];
            in_proj[l] = (const float*)d_in[b + 6];
            ln_b[l] = (const float*)d_in[b + 7];    ln_w[l] = (const float*)d_in[b + 8];
            out_proj[l] = (const float*)d_in[b + 9]; x_proj[l] = (const float*)d_in[b + 10];
        }
        mlp_w1 = (const float*)d_in[45]; mlp_w2 = (const float*)d_in[46];
        norm_b = (const float*)d_in[47]; norm_w = (const float*)d_in[48];
    }

    // ---- workspace layout (with temporal reuse; ~90 MB total) ----
    char* wsb = (char*)d_ws;
    size_t off = 0;
    auto alloc = [&](size_t bytes) -> char* {
        size_t a = (off + 255) & ~(size_t)255; off = a + bytes; return wsb + a;
    };
    h16_t* pool = (h16_t*)alloc((size_t)573440 * 2);              // f16 weights
    float* xs   = (float*)alloc((size_t)NTOK * 128 * 4);          // gathered tokens
    h16_t* h16  = (h16_t*)alloc((size_t)NTOK * 128 * 2);          // LN out / later yb16
    h16_t* xz16 = (h16_t*)alloc((size_t)NTOK * 512 * 2);          // xz / later hbuf+yfin
    h16_t* x16  = (h16_t*)alloc((size_t)NTOK * 512 * 2);          // x (lo), y (hi) / hid16
    float* xdbl = (float*)alloc((size_t)NTOK * 48 * 4);           // dt|B|C projections
    float* chA  = (float*)alloc((size_t)NC * 4096 * 4);
    float* chB  = (float*)alloc((size_t)NC * 4096 * 4);
    float* hini = (float*)alloc((size_t)NC * 4096 * 4);
    float* ybl  = (float*)alloc((size_t)NTOK * 128 * 4);          // block output (resid)
    h16_t* y16  = x16 + (size_t)NTOK * 256;
    h16_t* yb16 = h16;                                            // reuse
    h16_t* hid16 = x16;                                           // reuse (x16+y16 region)
    float* hbuf = (float*)xz16;                                   // reuse
    float* yfin = (float*)((char*)xz16 + (size_t)NTOK * 128 * 4); // reuse
    (void)ws_size;

    h16_t *ip16[4], *xp16[4], *op16[4];
    for (int l = 0; l < 4; l++) {
        ip16[l] = pool + (size_t)l * 110592;
        xp16[l] = ip16[l] + 65536;
        op16[l] = ip16[l] + 77824;
    }
    h16_t* m1_16 = pool + 442368;
    h16_t* m2_16 = pool + 507904;

    // ---- weight conversion ----
    for (int l = 0; l < 4; l++) {
        k_cvt<<<(512 * 128 + 255) / 256, 256, 0, stream>>>(in_proj[l], ip16[l], 512, 128, 512);
        k_cvt<<<(48 * 256 + 255) / 256, 256, 0, stream>>>(x_proj[l], xp16[l], 48, 256, 40);
        k_cvt<<<(128 * 256 + 255) / 256, 256, 0, stream>>>(out_proj[l], op16[l], 128, 256, 128);
    }
    k_cvt<<<(512 * 128 + 255) / 256, 256, 0, stream>>>(mlp_w1, m1_16, 512, 128, 512);
    k_cvt<<<(128 * 512 + 255) / 256, 256, 0, stream>>>(mlp_w2, m2_16, 128, 512, 128);

    // ---- gather (efficient_scan flat reinterpretation) ----
    k_gather<<<(8 * 128 * LSEQ) / 256, 256, 0, stream>>>(feature, xs);

    // ---- per-sequence Mamba blocks ----
    for (int sq = 0; sq < 8; sq++) {
        int d = sq & 3;
        size_t tb = (size_t)sq * LSEQ;
        k_ln<<<(LSEQ + 3) / 4, 128, 0, stream>>>(xs + tb * 128, ln_w[d], ln_b[d],
                                                 h16 + tb * 128, LSEQ);
        // GEMM1: [3072,128] @ in_proj^T -> xz [3072,512] (f16)
        k_gemm<4, false, true, false, 0><<<dim3(LSEQ / 16, 8), 32, 0, stream>>>(
            h16 + tb * 128, ip16[d], LSEQ, 128,
            nullptr, 0, xz16 + tb * 512, 512, nullptr, 0);
        k_conv<<<LSEQ, 256, 0, stream>>>(xz16, conv_w[d], conv_b[d], x16, sq);
        // GEMM2: [3072,256] @ x_proj^T(pad 48) -> xdbl [3072,48] (f32)
        k_gemm<3, true, false, false, 0><<<dim3(LSEQ / 16, 1), 32, 0, stream>>>(
            x16 + tb * 256, xp16[d], LSEQ, 256,
            xdbl + tb * 48, 48, nullptr, 0, nullptr, 0);
        // 3-pass chunked selective scan
        k_scanA<<<NC, 256, 0, stream>>>(xdbl, x16, A_log[d], dt_w[d], dt_b[d], chA, chB, sq);
        k_scanB<<<16, 256, 0, stream>>>(chA, chB, hini);
        k_scanC<<<NC, 256, 0, stream>>>(xdbl, x16, A_log[d], dt_w[d], dt_b[d], hini, y16, sq);
        k_post<<<LSEQ, 256, 0, stream>>>(y16, x16, xz16, Dp[d], sq);
        // GEMM3: [3072,256] @ out_proj^T + resid -> y_blocks f32 + yb16 f16
        k_gemm<4, true, true, true, 0><<<dim3(LSEQ / 16, 2), 32, 0, stream>>>(
            y16 + tb * 256, op16[d], LSEQ, 256,
            ybl + tb * 128, 128, yb16 + tb * 128, 128, xs + tb * 128, 128);
    }

    // ---- shared MLP + final LN residual ----
    k_gemm<4, false, true, false, 1><<<dim3(NTOK / 16, 8), 32, 0, stream>>>(   // GELU fused
        yb16, m1_16, NTOK, 128, nullptr, 0, hid16, 512, nullptr, 0);
    k_gemm<4, true, false, false, 0><<<dim3(NTOK / 16, 2), 32, 0, stream>>>(
        hid16, m2_16, NTOK, 512, hbuf, 128, nullptr, 0, nullptr, 0);
    k_lnres<<<(NTOK + 3) / 4, 128, 0, stream>>>(hbuf, norm_w, norm_b, ybl, yfin, NTOK);

    // ---- merge back to [2,128,96,128] ----
    k_merge<<<(8 * 128 * LSEQ) / 256, 256, 0, stream>>>(yfin, (float*)d_out);
}
